// MixtralAttention_16733192585649
// MI455X (gfx1250) — matrix operand order repaired and checked by reading, not compiled, not run on hardware
//
#include <hip/hip_runtime.h>
#include <hip/hip_bf16.h>
#include <math.h>

typedef __attribute__((ext_vector_type(16))) __bf16 v16bf;
typedef __attribute__((ext_vector_type(8)))  float  v8f;

union FragBF { v16bf v; uint4 u[2]; };

#define WMMA_BF16(a, b, c) \
  __builtin_amdgcn_wmma_f32_16x16x32_bf16(false, (a), false, (b), (short)0, (c), false, false)

// Async global->LDS copy (CDNA5 GLOBAL_LOAD_ASYNC_TO_LDS_B128, ASYNCcnt).
// VDST = per-lane LDS byte offset, VADDR = 64-bit global address.
#define ASYNC_G2L_B128(ldsoff, gaddr) \
  asm volatile("global_load_async_to_lds_b128 %0, %1, off" \
               :: "v"(ldsoff), "v"(gaddr) : "memory")
#define WAIT_ASYNC0() asm volatile("s_wait_asynccnt 0" ::: "memory")

// A-operand 16-bit layout (ISA 7.12.2): lanes 0-15 hold K[0..7],K[16..23];
// lanes 16-31 hold K[8..15],K[24..31].
__device__ __forceinline__ v16bf load_afrag(const __bf16* __restrict__ base,
                                            int ld, int row, int kb, int lane) {
  int k0 = kb + ((lane >> 4) << 3);
  const __bf16* p = base + (size_t)row * ld + k0;
  FragBF f;
  f.u[0] = *(const uint4*)(p);
  f.u[1] = *(const uint4*)(p + 16);
  return f.v;
}

// B-operand 16-bit layout: lanes 0-15 hold K[0..7],K[16..23]; lanes 16-31 hold K[8..15],K[24..31].
__device__ __forceinline__ v16bf load_bfrag(const __bf16* __restrict__ base,
                                            int ld, int row, int kb, int lane) {
  int k0 = kb + ((lane >> 4) << 3);
  const __bf16* p = base + (size_t)row * ld + k0;
  FragBF f;
  f.u[0] = *(const uint4*)(p);
  f.u[1] = *(const uint4*)(p + 16);
  return f.v;
}

// ---------------------------------------------------------------------------
// f32 -> bf16 conversion
// ---------------------------------------------------------------------------
__global__ void cvt_bf16_kernel(const float* __restrict__ in,
                                __bf16* __restrict__ out, size_t n) {
  size_t i = (size_t)blockIdx.x * blockDim.x + threadIdx.x;
  size_t stride = (size_t)gridDim.x * blockDim.x;
  for (; i < n; i += stride) out[i] = (__bf16)in[i];
}

// ---------------------------------------------------------------------------
// GEMM: C[M,N] = A[M,K] (bf16 row-major) * W[N,K]^T (bf16 rows=N)
// Block 64(M) x 128(N), 256 threads = 8 waves (2x4), wave tile 32x32.
// K staged in 64-deep double-buffered LDS tiles via async global->LDS DMA:
// 8 WMMAs per barrier, 24KB DMA per stage overlapped with compute.
// mode 0: f32 row-major out.  mode 1: bf16 head-major out [n/128][M][128].
// ---------------------------------------------------------------------------
__global__ __launch_bounds__(256) void gemm_bf16_kernel(
    const __bf16* __restrict__ A, const __bf16* __restrict__ W,
    float* __restrict__ outF, __bf16* __restrict__ outH,
    int M, int N, int K, int mode) {
  // Per buffer: A tile 64x64 (8KB) then B tile 128x64 (16KB).
  __shared__ __bf16 smem[2][(64 + 128) * 64];

  const int tid  = threadIdx.x;
  const int lane = tid & 31;
  const int w    = tid >> 5;
  const int wm   = w & 1;
  const int wn   = w >> 1;
  const int m_base = blockIdx.y * 64 + wm * 32;
  const int n_base = blockIdx.x * 128 + wn * 32;
  const int lm = wm * 32;           // local tile coords in LDS
  const int ln = wn * 32;

  // Staging: 16B chunks, row-major ld=64. A: 512 chunks (2/thread);
  // B: 1024 chunks (4/thread).
  const int mA0 = tid >> 3;               // chunks tid, tid+256
  const int mA1 = (tid + 256) >> 3;
  const int nB0 = tid >> 3;               // chunks tid, +256, +512, +768
  const int nB1 = (tid + 256) >> 3;
  const int nB2 = (tid + 512) >> 3;
  const int nB3 = (tid + 768) >> 3;
  const int kC  = (tid & 7) * 8;          // same k-subchunk for all
  const __bf16* gA0 = A + (size_t)(blockIdx.y * 64  + mA0) * K + kC;
  const __bf16* gA1 = A + (size_t)(blockIdx.y * 64  + mA1) * K + kC;
  const __bf16* gB0 = W + (size_t)(blockIdx.x * 128 + nB0) * K + kC;
  const __bf16* gB1 = W + (size_t)(blockIdx.x * 128 + nB1) * K + kC;
  const __bf16* gB2 = W + (size_t)(blockIdx.x * 128 + nB2) * K + kC;
  const __bf16* gB3 = W + (size_t)(blockIdx.x * 128 + nB3) * K + kC;

  auto stage = [&](int buf, int kb) {
    __bf16* As = &smem[buf][0];
    __bf16* Bs = &smem[buf][64 * 64];
    ASYNC_G2L_B128((unsigned)(size_t)&As[mA0 * 64 + kC], gA0 + kb);
    ASYNC_G2L_B128((unsigned)(size_t)&As[mA1 * 64 + kC], gA1 + kb);
    ASYNC_G2L_B128((unsigned)(size_t)&Bs[nB0 * 64 + kC], gB0 + kb);
    ASYNC_G2L_B128((unsigned)(size_t)&Bs[nB1 * 64 + kC], gB1 + kb);
    ASYNC_G2L_B128((unsigned)(size_t)&Bs[nB2 * 64 + kC], gB2 + kb);
    ASYNC_G2L_B128((unsigned)(size_t)&Bs[nB3 * 64 + kC], gB3 + kb);
  };

  v8f c[2][2];
  c[0][0] = (v8f){}; c[0][1] = (v8f){}; c[1][0] = (v8f){}; c[1][1] = (v8f){};

  stage(0, 0);
  int buf = 0;
  for (int kb = 0; kb < K; kb += 64) {
    WAIT_ASYNC0();        // own async stores to LDS landed
    __syncthreads();      // everyone's landed; prior compute done reading
    if (kb + 64 < K) stage(buf ^ 1, kb + 64);

    const __bf16* As = &smem[buf][0];
    const __bf16* Bs = &smem[buf][64 * 64];

#pragma unroll
    for (int ks = 0; ks < 64; ks += 32) {
      FragBF a0, a1, b0, b1;
      const int ka = ks + ((lane >> 4) << 3);
      const __bf16* pa0 = As + (lm + (lane & 15)) * 64 + ka;
      a0.u[0] = *(const uint4*)(pa0);
      a0.u[1] = *(const uint4*)(pa0 + 16);
      const __bf16* pa1 = pa0 + 16 * 64;
      a1.u[0] = *(const uint4*)(pa1);
      a1.u[1] = *(const uint4*)(pa1 + 16);
      const int kk = ks + ((lane >> 4) << 3);
      const __bf16* pb0 = Bs + (ln + (lane & 15)) * 64 + kk;
      b0.u[0] = *(const uint4*)(pb0);
      b0.u[1] = *(const uint4*)(pb0 + 16);
      const __bf16* pb1 = pb0 + 16 * 64;
      b1.u[0] = *(const uint4*)(pb1);
      b1.u[1] = *(const uint4*)(pb1 + 16);
      c[0][0] = WMMA_BF16(a0.v, b0.v, c[0][0]);
      c[0][1] = WMMA_BF16(a0.v, b1.v, c[0][1]);
      c[1][0] = WMMA_BF16(a1.v, b0.v, c[1][0]);
      c[1][1] = WMMA_BF16(a1.v, b1.v, c[1][1]);
    }
    buf ^= 1;
  }

  const int rofs = (lane >> 4) << 3;
#pragma unroll
  for (int mi = 0; mi < 2; ++mi)
#pragma unroll
    for (int ni = 0; ni < 2; ++ni)
#pragma unroll
      for (int r = 0; r < 8; ++r) {
        int m = m_base + mi * 16 + r + rofs;
        int n = n_base + ni * 16 + (lane & 15);
        float val = c[mi][ni][r];
        if (mode == 0) {
          outF[(size_t)m * N + n] = val;
        } else {
          outH[((size_t)(n >> 7) * M + m) * 128 + (n & 127)] = (__bf16)val;
        }
      }
}

// ---------------------------------------------------------------------------
// In-place RoPE on bf16 head-major buffer [nheads][T][128]; scale folded in.
// ---------------------------------------------------------------------------
__global__ void rope_bf16_kernel(__bf16* __restrict__ buf,
                                 const float* __restrict__ cosT,
                                 const float* __restrict__ sinT,
                                 int nheads, int T, float scale) {
  size_t idx = (size_t)blockIdx.x * blockDim.x + threadIdx.x;
  size_t total = (size_t)nheads * T * 64;
  if (idx >= total) return;
  int i = (int)(idx & 63);
  size_t ht = idx >> 6;
  int t = (int)(ht % T);
  __bf16* p = buf + ht * 128 + 2 * i;
  float x1 = (float)p[0];
  float x2 = (float)p[1];
  float c = cosT[(size_t)t * 64 + i];
  float s = sinT[(size_t)t * 64 + i];
  p[0] = (__bf16)((x1 * c - x2 * s) * scale);
  p[1] = (__bf16)((x1 * s + x2 * c) * scale);
}

// ---------------------------------------------------------------------------
// Sliding-window flash attention.
// Q,K,V head-major bf16 [h][T][128]; Q pre-scaled by 1/sqrt(D).
// Grid: (T/64, nh), 128 threads = 4 waves, wave owns 16 query rows.
// Output: bf16 row-major [T][nh*128].
// ---------------------------------------------------------------------------
__global__ __launch_bounds__(128) void attn_kernel(
    const __bf16* __restrict__ Qg, const __bf16* __restrict__ Kg,
    const __bf16* __restrict__ Vg, __bf16* __restrict__ Oa,
    const int* __restrict__ win_ptr, int T, int nh, int nkv) {
  __shared__ __bf16 Vt[128][32];        // transposed V tile: [dim][key]
  __shared__ __bf16 Pl[4][16][32];      // per-wave P tile

  const int window = *win_ptr;
  const int tid  = threadIdx.x;
  const int lane = tid & 31;
  const int w    = tid >> 5;
  const int q0   = blockIdx.x * 64;
  const int h    = blockIdx.y;
  const int n_rep = nh / nkv;
  const int kvh  = h / n_rep;

  const __bf16* Qh = Qg + (size_t)h   * T * 128;
  const __bf16* Kh = Kg + (size_t)kvh * T * 128;
  const __bf16* Vh = Vg + (size_t)kvh * T * 128;

  const int wq0 = q0 + w * 16;
  const int rofs = (lane >> 4) << 3;

  v16bf qa[4];
#pragma unroll
  for (int dk = 0; dk < 4; ++dk)
    qa[dk] = load_afrag(Qh, 128, wq0 + (lane & 15), dk * 32, lane);

  v8f o[8];
#pragma unroll
  for (int nt = 0; nt < 8; ++nt) o[nt] = (v8f){};
  float mrow[8], lrow[8];
#pragma unroll
  for (int r = 0; r < 8; ++r) { mrow[r] = -1e30f; lrow[r] = 0.0f; }

  int kb_lo = q0 - (window - 1);
  if (kb_lo < 0) kb_lo = 0;
  kb_lo &= ~31;

  for (int kb = kb_lo; kb < q0 + 64; kb += 32) {
    __syncthreads();
    for (int idx = tid; idx < 32 * 128; idx += 128) {
      int s = idx >> 7;
      int d = idx & 127;
      Vt[d][s] = Vh[(size_t)(kb + s) * 128 + d];
    }
    __syncthreads();

    // Wave-uniform skip keeps EXEC all-ones around WMMA.
    bool active = (kb <= wq0 + 15) && (kb + 31 >= wq0 - (window - 1));
    if (active) {
      v8f s0 = (v8f){}, s1 = (v8f){};
#pragma unroll
      for (int dk = 0; dk < 4; ++dk) {
        v16bf b0 = load_bfrag(Kh, 128, kb +      (lane & 15), dk * 32, lane);
        v16bf b1 = load_bfrag(Kh, 128, kb + 16 + (lane & 15), dk * 32, lane);
        s0 = WMMA_BF16(qa[dk], b0, s0);
        s1 = WMMA_BF16(qa[dk], b1, s1);
      }

      const int sc0 = kb + (lane & 15);
      const int sc1 = sc0 + 16;
#pragma unroll
      for (int r = 0; r < 8; ++r) {
        const int qi = wq0 + r + rofs;
        float x0 = s0[r];
        float x1 = s1[r];
        if (sc0 > qi || sc0 + window <= qi) x0 = -1e30f;
        if (sc1 > qi || sc1 + window <= qi) x1 = -1e30f;
        float mx = fmaxf(x0, x1);
        mx = fmaxf(mx, __shfl_xor(mx, 1));
        mx = fmaxf(mx, __shfl_xor(mx, 2));
        mx = fmaxf(mx, __shfl_xor(mx, 4));
        mx = fmaxf(mx, __shfl_xor(mx, 8));
        float mnew = fmaxf(mrow[r], mx);
        float alpha = __expf(mrow[r] - mnew);
        float p0 = __expf(x0 - mnew);
        float p1 = __expf(x1 - mnew);
        float rs = p0 + p1;
        rs += __shfl_xor(rs, 1);
        rs += __shfl_xor(rs, 2);
        rs += __shfl_xor(rs, 4);
        rs += __shfl_xor(rs, 8);
        lrow[r] = lrow[r] * alpha + rs;
        mrow[r] = mnew;
#pragma unroll
        for (int nt = 0; nt < 8; ++nt) o[nt][r] *= alpha;
        const int prow = r + rofs;
        Pl[w][prow][lane & 15]        = (__bf16)p0;
        Pl[w][prow][16 + (lane & 15)] = (__bf16)p1;
      }

      FragBF pf;
      {
        const int k0 = (lane >> 4) << 3;
        pf.u[0] = *(const uint4*)&Pl[w][lane & 15][k0];
        pf.u[1] = *(const uint4*)&Pl[w][lane & 15][k0 + 16];
      }
#pragma unroll
      for (int nt = 0; nt < 8; ++nt) {
        FragBF vf;
        const int d  = nt * 16 + (lane & 15);
        const int k0 = (lane >> 4) << 3;
        vf.u[0] = *(const uint4*)&Vt[d][k0];
        vf.u[1] = *(const uint4*)&Vt[d][k0 + 16];
        o[nt] = WMMA_BF16(pf.v, vf.v, o[nt]);
      }
    }
  }

  const int ostride = nh * 128;
#pragma unroll
  for (int r = 0; r < 8; ++r) {
    float inv = 1.0f / lrow[r];
    const int qi = wq0 + r + rofs;
#pragma unroll
    for (int nt = 0; nt < 8; ++nt) {
      int col = h * 128 + nt * 16 + (lane & 15);
      Oa[(size_t)qi * ostride + col] = (__bf16)(o[nt][r] * inv);
    }
  }
}

// ---------------------------------------------------------------------------
// Host side
// ---------------------------------------------------------------------------
extern "C" void kernel_launch(void* const* d_in, const int* in_sizes, int n_in,
                              void* d_out, int out_size, void* d_ws, size_t ws_size,
                              hipStream_t stream) {
  (void)n_in; (void)out_size; (void)ws_size;
  const float* x    = (const float*)d_in[0];
  const float* cosT = (const float*)d_in[1];
  const float* sinT = (const float*)d_in[2];
  const float* Wq   = (const float*)d_in[3];
  const float* Wk   = (const float*)d_in[4];
  const float* Wv   = (const float*)d_in[5];
  const float* Wo   = (const float*)d_in[6];
  const int*  win   = (const int*)d_in[7];
  float* out = (float*)d_out;

  const int D   = 128;
  const int T   = in_sizes[1] / (D / 2);
  const int C   = in_sizes[0] / T;
  const int nh  = in_sizes[3] / C / D;
  const int nkv = in_sizes[4] / C / D;

  char* ws = (char*)d_ws;
  size_t off = 0;
  auto alloc = [&](size_t bytes) { char* p = ws + off; off += (bytes + 255) & ~(size_t)255; return p; };
  __bf16* xb   = (__bf16*)alloc((size_t)T * C * 2);
  __bf16* wqb  = (__bf16*)alloc((size_t)nh  * D * C * 2);
  __bf16* wkb  = (__bf16*)alloc((size_t)nkv * D * C * 2);
  __bf16* wvb  = (__bf16*)alloc((size_t)nkv * D * C * 2);
  __bf16* wob  = (__bf16*)alloc((size_t)C * C * 2);
  __bf16* qb   = (__bf16*)alloc((size_t)nh  * T * D * 2);
  __bf16* kb   = (__bf16*)alloc((size_t)nkv * T * D * 2);
  __bf16* vb   = (__bf16*)alloc((size_t)nkv * T * D * 2);
  __bf16* attn = (__bf16*)alloc((size_t)T * C * 2);

  const int CVT_B = 256;
  auto cvt = [&](const float* src, __bf16* dst, size_t n) {
    int blocks = (int)((n + CVT_B * 8 - 1) / (CVT_B * 8));
    if (blocks > 4096) blocks = 4096;
    cvt_bf16_kernel<<<blocks, CVT_B, 0, stream>>>(src, dst, n);
  };
  cvt(x,  xb,  (size_t)T * C);
  cvt(Wq, wqb, (size_t)nh  * D * C);
  cvt(Wk, wkb, (size_t)nkv * D * C);
  cvt(Wv, wvb, (size_t)nkv * D * C);
  cvt(Wo, wob, (size_t)C * C);

  {
    dim3 gq(nh * D / 128, T / 64);
    gemm_bf16_kernel<<<gq, 256, 0, stream>>>(xb, wqb, nullptr, qb, T, nh * D, C, 1);
    dim3 gk(nkv * D / 128, T / 64);
    gemm_bf16_kernel<<<gk, 256, 0, stream>>>(xb, wkb, nullptr, kb, T, nkv * D, C, 1);
    gemm_bf16_kernel<<<gk, 256, 0, stream>>>(xb, wvb, nullptr, vb, T, nkv * D, C, 1);
  }

  {
    float inv_sqrt_d = 1.0f / sqrtf((float)D);
    size_t tq = (size_t)nh * T * (D / 2);
    size_t tk = (size_t)nkv * T * (D / 2);
    rope_bf16_kernel<<<(unsigned)((tq + 255) / 256), 256, 0, stream>>>(qb, cosT, sinT, nh, T, inv_sqrt_d);
    rope_bf16_kernel<<<(unsigned)((tk + 255) / 256), 256, 0, stream>>>(kb, cosT, sinT, nkv, T, 1.0f);
  }

  {
    dim3 ga(T / 64, nh);
    attn_kernel<<<ga, 128, 0, stream>>>(qb, kb, vb, attn, win, T, nh, nkv);
  }

  {
    dim3 go(C / 128, T / 64);
    gemm_bf16_kernel<<<go, 256, 0, stream>>>(attn, wob, out, nullptr, T, C, C, 0);
  }
}
